// CorrNet_54898271977903
// MI455X (gfx1250) — compile-verified, matrix operand order checked by reading, not run on hardware
//
#include <hip/hip_runtime.h>
#include <math.h>

#ifndef M_PI
#define M_PI 3.14159265358979323846
#endif

typedef __attribute__((ext_vector_type(16))) __bf16 v16bf;
typedef __attribute__((ext_vector_type(2)))  __bf16 v2bf;
typedef __attribute__((ext_vector_type(8)))  float  v8f;

#define MDIM 128
#define KDIM 64
#define TDIM 64
#define DIN  320
#define NWAVES 8

// ---- constants matching the reference ----
#define INV_SQRT_M_C 0.08838834764831845   // 1/sqrt(128)
#define INV_SQRT_K_C 0.125                 // 1/sqrt(64)
#define INV_SQRT_3_C 0.5773502691896258
#define TP_NORM_C    0.011048543456039806f // 1/sqrt(2*64*64)

// ---- fragmentized-weight layout (bf16 elements), both in d_ws and in LDS ----
// A-fragments of W^T (scale pre-folded): frag(mt,kt) -> 32 lanes x 16 bf16 (1KB)
#define OFF_W01 0        // w0_1 * 1/sqrt(M): 12mt x 4kt = 48 frags
#define OFF_W11 24576    // w1_1 * 1/sqrt(K)/C_RELU:  8 frags
#define OFF_W02 28672    // w0_2 * 1/sqrt(M): 48 frags
#define OFF_W12 53248    // w1_2 * 1/sqrt(K)/C_RELU:  8 frags
#define OFF_W0O 57344    // w0_o * 1/sqrt(M): 16 frags
#define OFF_W1O 65536    // w1_o * 1/sqrt(K):  8 frags
#define OFF_TP0 69632    // w_tp0:             8 frags
#define OFF_TP1 73728    // w_tp1 * 1/sqrt(3): 8 frags
#define WS_ELEMS 77824   // total bf16 elements (152 KB)

// Packed f32 -> 2x bf16 conversion (RNE) via v_cvt_pk_bf16_f32.
__device__ __forceinline__ v2bf cvt2(float a, float b) {
#if __has_builtin(__builtin_amdgcn_cvt_pk_bf16_f32)
  return __builtin_amdgcn_cvt_pk_bf16_f32(a, b);
#else
  v2bf r; r[0] = (__bf16)a; r[1] = (__bf16)b; return r;
#endif
}

__device__ __forceinline__ v16bf load_frag(const __bf16* p, int lane) {
  return *(const v16bf*)(p + (lane << 4));           // 32B per lane (ds_load_b128 x2)
}

// Build a bf16 B-fragment (32 x 16) from two stacked f32 D tiles.
// B layout: lane L holds col n=L%16; element e -> k=(e&7)+16*(e>>3)+8*(L>>4).
// D tile:   lane L holds col n=L%16; vgpr r -> row m=r+8*(L>>4).
__device__ __forceinline__ v16bf make_b(v8f lo, v8f hi) {
  union { v16bf v; v2bf p[8]; } u;
#pragma unroll
  for (int j = 0; j < 4; ++j) u.p[j] = cvt2(lo[2 * j], lo[2 * j + 1]);
#pragma unroll
  for (int j = 0; j < 4; ++j) u.p[4 + j] = cvt2(hi[2 * j], hi[2 * j + 1]);
  return u.v;
}

__device__ __forceinline__ v8f wmma_bf16(v16bf a, v16bf b, v8f c) {
  return __builtin_amdgcn_wmma_f32_16x16x32_bf16(false, a, false, b,
                                                 (short)0, c, false, false);
}

// Per-feature bias as WMMA C operand: C[r] = b[f0 + r] (row-broadcast in D layout).
__device__ __forceinline__ v8f load_bias(const float* __restrict__ b, int f0) {
  v8f c;
#pragma unroll
  for (int r = 0; r < 8; ++r) c[r] = b[f0 + r];
  return c;
}

// ---- prep: swizzle fp32 weight (Kin rows x Mout cols) into bf16 A-fragments of
// scale*W^T.  A layout: lane L row m=L%16; element e -> k=(e&7)+16*(e>>3)+8*(L>>4).
__global__ void fragw_kernel(const float* __restrict__ W, __bf16* __restrict__ dst,
                             int Kin, int Mout, float scale) {
  const int KT   = Kin >> 5;
  const int frag = blockIdx.x;
  const int mt   = frag / KT;
  const int kt   = frag % KT;
  const int t = threadIdx.x;          // 512 threads: (lane, elem)
  const int L = t >> 4;
  const int e = t & 15;
  const int m = 16 * mt + (L & 15);
  const int k = 32 * kt + (e & 7) + 16 * (e >> 3) + 8 * (L >> 4);
  dst[(size_t)frag * 512 + L * 16 + e] = (__bf16)(W[(size_t)k * Mout + m] * scale);
}

// ---- main fused kernel: 1 wave == 16 samples, all state in registers;
//      all weight fragments staged once per workgroup in LDS. ----
__global__ __launch_bounds__(32 * NWAVES) void corrnet_main(
    const float* __restrict__ x, const float* __restrict__ shift,
    const float* __restrict__ oscale,
    const float* __restrict__ b0_1, const float* __restrict__ b0_2,
    const float* __restrict__ b0_o,
    const __bf16* __restrict__ ws,
    float* __restrict__ out, int n, float inv_cs) {
  __shared__ __bf16 wlds[WS_ELEMS];            // 152 KB: all fragments, WG-shared

  // Cooperative global->LDS fill (uniform across the block, before any early-out).
  {
    const uint4* src = (const uint4*)ws;
    uint4* dst = (uint4*)wlds;
#pragma unroll 2
    for (int idx = threadIdx.x; idx < WS_ELEMS / 8; idx += 32 * NWAVES)
      dst[idx] = src[idx];
  }
  __syncthreads();

  const int lane = threadIdx.x & 31;
  const int wave = threadIdx.x >> 5;
  const int half = lane >> 4;
  const int col  = lane & 15;
  const int base = (blockIdx.x * NWAVES + wave) << 4;
  if (base >= n) return;                        // wave-uniform: EXEC stays all-ones
  const int sample = base + col;
  const float* xr = x + (size_t)sample * DIN;

  // Residual state, transposed, in D layout (fp32): features on rows, samples on lanes.
  v8f xsD[8];
#pragma unroll
  for (int mt = 0; mt < 8; ++mt) {
    const int f0 = 16 * mt + 8 * half;
#pragma unroll
    for (int r = 0; r < 8; ++r) xsD[mt][r] = xr[f0 + r] - shift[f0 + r];
  }
  v8f xvD[3][4];
#pragma unroll
  for (int mt = 0; mt < 4; ++mt) {
    const int g0 = MDIM + 3 * (16 * mt + 8 * half);
    float buf[24];
#pragma unroll
    for (int t = 0; t < 24; ++t) buf[t] = xr[g0 + t] - shift[g0 + t];
#pragma unroll
    for (int r = 0; r < 8; ++r)
#pragma unroll
      for (int i = 0; i < 3; ++i) xvD[i][mt][r] = buf[3 * r + i];
  }

  const __bf16* w0f[2] = { wlds + OFF_W01, wlds + OFF_W02 };
  const __bf16* w1f[2] = { wlds + OFF_W11, wlds + OFF_W12 };
  const float*  b0p[2] = { b0_1, b0_2 };

#pragma unroll
  for (int blk = 0; blk < 2; ++blk) {
    // Snapshot xs as bf16 B-fragments (live only through GEMM1).
    v16bf xsB[4];
#pragma unroll
    for (int kt = 0; kt < 4; ++kt) xsB[kt] = make_b(xsD[2 * kt], xsD[2 * kt + 1]);

    // GEMM1: ys^T = (w0^T/sqrt(M)) @ xs^T + b0  (scale folded, bias as C operand)
    v8f g[4];                                   // bare ReLU gates (1/C_RELU folded into w1)
#pragma unroll
    for (int mt = 0; mt < 12; ++mt) {
      v8f acc = load_bias(b0p[blk], 16 * mt + 8 * half);
#pragma unroll
      for (int kt = 0; kt < 4; ++kt)
        acc = wmma_bf16(load_frag(w0f[blk] + (mt * 4 + kt) * 512, lane), xsB[kt], acc);
      if (mt < 8) {
#pragma unroll
        for (int r = 0; r < 8; ++r) {           // xs += silu(v)/C_SILU
          float v = acc[r];
          xsD[mt][r] += __fdividef(v * inv_cs, 1.f + __expf(-v));
        }
      } else {
#pragma unroll
        for (int r = 0; r < 8; ++r) g[mt - 8][r] = fmaxf(acc[r], 0.f);
      }
    }

    // GEMM2: xv += (w1^T/sqrt(K)/C_RELU @ xv^T) * relu(ys)   -> single fma per element
    v16bf xvB[3][2];
#pragma unroll
    for (int i = 0; i < 3; ++i)
#pragma unroll
      for (int kt = 0; kt < 2; ++kt) xvB[i][kt] = make_b(xvD[i][2 * kt], xvD[i][2 * kt + 1]);
#pragma unroll
    for (int mt = 0; mt < 4; ++mt) {
      v16bf wfr[2];
#pragma unroll
      for (int kt = 0; kt < 2; ++kt) wfr[kt] = load_frag(w1f[blk] + (mt * 2 + kt) * 512, lane);
#pragma unroll
      for (int i = 0; i < 3; ++i) {
        v8f acc = {};
#pragma unroll
        for (int kt = 0; kt < 2; ++kt) acc = wmma_bf16(wfr[kt], xvB[i][kt], acc);
#pragma unroll
        for (int r = 0; r < 8; ++r)
          xvD[i][mt][r] = fmaf(acc[r], g[mt][r], xvD[i][mt][r]);
      }
    }
  }

  // ---- output head (all scales folded into fragments; biases via C) ----
  v16bf osB[4];
#pragma unroll
  for (int kt = 0; kt < 4; ++kt) osB[kt] = make_b(xsD[2 * kt], xsD[2 * kt + 1]);
  v16bf ovB[3][2];
#pragma unroll
  for (int i = 0; i < 3; ++i)
#pragma unroll
    for (int kt = 0; kt < 2; ++kt) ovB[i][kt] = make_b(xvD[i][2 * kt], xvD[i][2 * kt + 1]);
  // xsD / xvD dead from here on.

  v8f zs[4];
#pragma unroll
  for (int mt = 0; mt < 4; ++mt) {
    v8f acc = load_bias(b0_o, 16 * mt + 8 * half);
#pragma unroll
    for (int kt = 0; kt < 4; ++kt)
      acc = wmma_bf16(load_frag(wlds + OFF_W0O + (mt * 4 + kt) * 512, lane), osB[kt], acc);
    zs[mt] = acc;                               // zero epilogue
  }
  v8f zv[3][4];
#pragma unroll
  for (int mt = 0; mt < 4; ++mt) {
    v16bf wfr[2];
#pragma unroll
    for (int kt = 0; kt < 2; ++kt) wfr[kt] = load_frag(wlds + OFF_W1O + (mt * 2 + kt) * 512, lane);
#pragma unroll
    for (int i = 0; i < 3; ++i) {
      v8f acc = {};
#pragma unroll
      for (int kt = 0; kt < 2; ++kt) acc = wmma_bf16(wfr[kt], ovB[i][kt], acc);
      zv[i][mt] = acc;                          // zero epilogue (1/sqrt(K) folded)
    }
  }

  // Quadratic forms: y0 = sum_j (w_tp0^T zs^T)[j,n] * zs^T[j,n]; same per zv channel.
  v16bf zsB[2] = { make_b(zs[0], zs[1]), make_b(zs[2], zs[3]) };
  float part0 = 0.f;
#pragma unroll
  for (int mt = 0; mt < 4; ++mt) {
    v8f u = {};
#pragma unroll
    for (int kt = 0; kt < 2; ++kt)
      u = wmma_bf16(load_frag(wlds + OFF_TP0 + (mt * 2 + kt) * 512, lane), zsB[kt], u);
#pragma unroll
    for (int r = 0; r < 8; ++r) part0 = fmaf(u[r], zs[mt][r], part0);
  }

  v16bf zvB[3][2];
#pragma unroll
  for (int i = 0; i < 3; ++i)
#pragma unroll
    for (int kt = 0; kt < 2; ++kt) zvB[i][kt] = make_b(zv[i][2 * kt], zv[i][2 * kt + 1]);
  float part1 = 0.f;                            // 1/sqrt(3) folded into w_tp1 frags
#pragma unroll
  for (int mt = 0; mt < 4; ++mt) {
    v16bf wfr[2];
#pragma unroll
    for (int kt = 0; kt < 2; ++kt) wfr[kt] = load_frag(wlds + OFF_TP1 + (mt * 2 + kt) * 512, lane);
#pragma unroll
    for (int i = 0; i < 3; ++i) {
      v8f u = {};
#pragma unroll
      for (int kt = 0; kt < 2; ++kt) u = wmma_bf16(wfr[kt], zvB[i][kt], u);
#pragma unroll
      for (int r = 0; r < 8; ++r) part1 = fmaf(u[r], zv[i][mt][r], part1);
    }
  }

  float tot = part0 + part1;
  tot += __shfl_xor(tot, 16, 32);               // combine the two feature halves
  float y = tot * TP_NORM_C / oscale[0];
  if (half == 0) out[sample] = y;
}

// ---- host-side activation norm constants (matches reference trapezoid) ----
static double silu_d(double z) { return z / (1.0 + exp(-z)); }
static double relu_d(double z) { return z > 0.0 ? z : 0.0; }
static double act_norm(double (*f)(double)) {
  const int n = 20001; const double a = -8.0, b = 8.0, h = (b - a) / (n - 1);
  double sum = 0.0, prev = 0.0;
  for (int i = 0; i < n; ++i) {
    double z = a + h * i;
    double pdf = exp(-0.5 * z * z) / sqrt(2.0 * M_PI);
    double v = f(z); v = v * v * pdf;
    if (i) sum += 0.5 * (v + prev) * h;
    prev = v;
  }
  return sqrt(sum);
}

extern "C" void kernel_launch(void* const* d_in, const int* in_sizes, int n_in,
                              void* d_out, int out_size, void* d_ws, size_t ws_size,
                              hipStream_t stream) {
  const float* x     = (const float*)d_in[0];
  const float* shift = (const float*)d_in[1];
  const float* oscl  = (const float*)d_in[2];
  const float* w0_1  = (const float*)d_in[3];
  const float* b0_1  = (const float*)d_in[4];
  const float* w1_1  = (const float*)d_in[5];
  const float* w0_2  = (const float*)d_in[6];
  const float* b0_2  = (const float*)d_in[7];
  const float* w1_2  = (const float*)d_in[8];
  const float* w0_o  = (const float*)d_in[9];
  const float* b0_o  = (const float*)d_in[10];
  const float* w1_o  = (const float*)d_in[11];
  const float* wtp0  = (const float*)d_in[12];
  const float* wtp1  = (const float*)d_in[13];
  __bf16* ws = (__bf16*)d_ws;
  const int n = in_sizes[0] / DIN;

  const double cS = act_norm(silu_d);
  const double cR = act_norm(relu_d);
  const float icS = (float)(1.0 / cS);
  const float s_w0 = (float)INV_SQRT_M_C;
  const float s_w1 = (float)(INV_SQRT_K_C / cR);   // fold 1/C_RELU into gate path
  const float s_wo = (float)INV_SQRT_K_C;
  const float s_t1 = (float)INV_SQRT_3_C;

  // Pre-swizzle weights into bf16 A-fragments (W^T tiles, scales folded) in d_ws.
  fragw_kernel<<<48, 512, 0, stream>>>(w0_1, ws + OFF_W01, 128, 192, s_w0);
  fragw_kernel<<< 8, 512, 0, stream>>>(w1_1, ws + OFF_W11,  64,  64, s_w1);
  fragw_kernel<<<48, 512, 0, stream>>>(w0_2, ws + OFF_W02, 128, 192, s_w0);
  fragw_kernel<<< 8, 512, 0, stream>>>(w1_2, ws + OFF_W12,  64,  64, s_w1);
  fragw_kernel<<<16, 512, 0, stream>>>(w0_o, ws + OFF_W0O, 128,  64, s_w0);
  fragw_kernel<<< 8, 512, 0, stream>>>(w1_o, ws + OFF_W1O,  64,  64, s_wo);
  fragw_kernel<<< 8, 512, 0, stream>>>(wtp0, ws + OFF_TP0,  64,  64, 1.0f);
  fragw_kernel<<< 8, 512, 0, stream>>>(wtp1, ws + OFF_TP1,  64,  64, s_t1);

  const int grid = (n + 16 * NWAVES - 1) / (16 * NWAVES);
  corrnet_main<<<grid, 32 * NWAVES, 0, stream>>>(
      x, shift, oscl, b0_1, b0_2, b0_o, ws, (float*)d_out, n, icS);
}
